// LSTMClassifier_17420387352750
// MI455X (gfx1250) — matrix operand order repaired and checked by reading, not compiled, not run on hardware
//
#include <hip/hip_runtime.h>

// ---------------------------------------------------------------------------
// Problem constants (match reference)
// ---------------------------------------------------------------------------
constexpr int B  = 128;
constexpr int T  = 256;
constexpr int IN = 256;
constexpr int H  = 512;
constexpr int NC = 10;
constexpr int G4H = 4 * H;   // 2048

typedef __attribute__((ext_vector_type(16))) __bf16 v16bf;
typedef __attribute__((ext_vector_type(8)))  __bf16 v8bf;
typedef __attribute__((ext_vector_type(8)))  float  v8f;
typedef __attribute__((ext_vector_type(4)))  int    v4i;

// CDNA5 async global->LDS path (ASYNCcnt-tracked). Guarded: falls back to
// plain load + ds_store if the builtin is absent on this toolchain.
#if defined(__HIP_DEVICE_COMPILE__) && __has_builtin(__builtin_amdgcn_global_load_async_to_lds_b128)
#define ASYNC_LDS 1
typedef __attribute__((address_space(1))) v4i* g_v4i_ptr;
typedef __attribute__((address_space(3))) v4i* l_v4i_ptr;
#else
#define ASYNC_LDS 0
#endif

// ---------------------------------------------------------------------------
// WMMA fragment helpers (contiguous k order, same in A and B, wave32)
// ---------------------------------------------------------------------------
__device__ __forceinline__ v16bf load_frag_a(const __bf16* __restrict__ base,
                                             int ld, int mBase, int kBase, int lane) {
    // A 16x32 bf16: lanes 0-15 -> M=0..15, K=0..15;
    // lanes 16-31 -> same M rows, K=16..31 (contiguous)
    const int m   = mBase + (lane & 15);
    const int off = (lane & 16) ? 16 : 0;
    const __bf16* p = base + (size_t)m * ld + kBase + off;
    v8bf lo = *(const v8bf*)(p);
    v8bf hi = *(const v8bf*)(p + 8);
    return __builtin_shufflevector(lo, hi, 0,1,2,3,4,5,6,7,8,9,10,11,12,13,14,15);
}

__device__ __forceinline__ v16bf load_frag_b(const __bf16* __restrict__ base,
                                             int ld, int nBase, int kBase, int lane) {
    // B 32x16 bf16: lane holds column n = lane&15; lanes 0-15: K=0..15,
    // lanes 16-31: K=16..31 (contiguous). W stored (N x K) row-major => direct load.
    const int n   = nBase + (lane & 15);
    const int off = (lane & 16) ? 16 : 0;
    const __bf16* p = base + (size_t)n * ld + kBase + off;
    v8bf lo = *(const v8bf*)(p);
    v8bf hi = *(const v8bf*)(p + 8);
    return __builtin_shufflevector(lo, hi, 0,1,2,3,4,5,6,7,8,9,10,11,12,13,14,15);
}

// B-fragment sourced from LDS (smem row-major, row stride = H elements)
__device__ __forceinline__ v16bf load_frag_b_lds(const __bf16* smem,
                                                 int rBase, int kBase, int lane) {
    const int r   = rBase + (lane & 15);
    const int off = (lane & 16) ? 16 : 0;
    const __bf16* p = smem + r * H + kBase + off;
    v8bf lo = *(const v8bf*)(p);
    v8bf hi = *(const v8bf*)(p + 8);
    return __builtin_shufflevector(lo, hi, 0,1,2,3,4,5,6,7,8,9,10,11,12,13,14,15);
}

__device__ __forceinline__ v8f wmma_bf16(v16bf a, v16bf b, v8f c) {
    return __builtin_amdgcn_wmma_f32_16x16x32_bf16(false, a, false, b, (short)0, c,
                                                   false, false);
}

__device__ __forceinline__ float sigmoid_(float x) {
    return 1.f / (1.f + __expf(-x));
}
__device__ __forceinline__ float tanh_(float x) {
    x = fminf(fmaxf(x, -15.f), 15.f);
    float e = __expf(-2.f * x);
    return (1.f - e) / (1.f + e);
}

// ---------------------------------------------------------------------------
// Elementwise helpers
// ---------------------------------------------------------------------------
__global__ void cast_f32_bf16(const float* __restrict__ src, __bf16* __restrict__ dst,
                              long n) {
    long i = (long)blockIdx.x * blockDim.x + threadIdx.x;
    long stride = (long)gridDim.x * blockDim.x;
    for (; i < n; i += stride) dst[i] = (__bf16)src[i];
}

__global__ void init_state(__bf16* __restrict__ h, float* __restrict__ c, int n) {
    int i = blockIdx.x * blockDim.x + threadIdx.x;
    if (i < n) { h[i] = (__bf16)0.f; c[i] = 0.f; }
}

// ---------------------------------------------------------------------------
// Input-projection GEMM: C[M,N] = A[M,K](bf16) * W[N,K]^T(bf16) + b0[N]+b1[N]
// Register-blocked: each wave computes a 16x64 strip (1 M-tile x 4 N-tiles),
// reusing one A-fragment across 4 WMMAs per K step.
// ---------------------------------------------------------------------------
__global__ __launch_bounds__(256) void gemm_bf16_f32(
    const __bf16* __restrict__ A, const __bf16* __restrict__ W,
    const float* __restrict__ bias0, const float* __restrict__ bias1,
    float* __restrict__ Cmat, int M, int N, int K)
{
    const int lane = threadIdx.x & 31;
    const int wave = threadIdx.x >> 5;
    const int grpN = N >> 6;                      // groups of 4 N-tiles
    const int numGrp = (M >> 4) * grpN;

    for (int grp = blockIdx.x * 8 + wave; grp < numGrp; grp += gridDim.x * 8) {
        const int mt = grp / grpN;
        const int ng = grp - mt * grpN;
        const int mBase = mt << 4, nBase = ng << 6;

        v8f acc0 = {0.f,0.f,0.f,0.f,0.f,0.f,0.f,0.f};
        v8f acc1 = acc0, acc2 = acc0, acc3 = acc0;
        for (int k = 0; k < K; k += 32) {
            v16bf a = load_frag_a(A, K, mBase, k, lane);
            acc0 = wmma_bf16(a, load_frag_b(W, K, nBase +  0, k, lane), acc0);
            acc1 = wmma_bf16(a, load_frag_b(W, K, nBase + 16, k, lane), acc1);
            acc2 = wmma_bf16(a, load_frag_b(W, K, nBase + 32, k, lane), acc2);
            acc3 = wmma_bf16(a, load_frag_b(W, K, nBase + 48, k, lane), acc3);
        }
        const int mOff = mBase + ((lane & 16) ? 8 : 0);
        v8f accs[4] = {acc0, acc1, acc2, acc3};
        #pragma unroll
        for (int s = 0; s < 4; ++s) {
            const int n = nBase + s * 16 + (lane & 15);
            const float bn = bias0[n] + bias1[n];
            #pragma unroll
            for (int i = 0; i < 8; ++i)
                Cmat[(size_t)(mOff + i) * N + n] = accs[s][i] + bn;
        }
    }
}

// ---------------------------------------------------------------------------
// Fused recurrent step: gates = xg[:,t,:] + h_in @ Whh^T ; LSTM pointwise.
// grid = H/16 blocks (16 hidden cols each), 8 waves (one 16-row M tile each).
// The block's 64 W_hh rows (4 gates x 16 cols, 64 KB) are staged in LDS once
// (async global->LDS when available) and reused by all 8 waves.
// ---------------------------------------------------------------------------
__global__ __launch_bounds__(256) void lstm_step(
    const float*  __restrict__ xg,      // [B][T][4H] f32 (input projection + biases)
    const __bf16* __restrict__ Whh,     // [4H][H] bf16
    const __bf16* __restrict__ h_in,    // [B][H] bf16
    __bf16*       __restrict__ h_out,   // [B][H] bf16
    float*        __restrict__ c_state, // [B][H] f32 (in/out)
    __bf16*       __restrict__ h_seq,   // [B][T][H] bf16 or nullptr
    int t)
{
    extern __shared__ __bf16 smem[];        // 64 rows x H bf16 = 64 KB

    const int lane  = threadIdx.x & 31;
    const int wave  = threadIdx.x >> 5;     // 0..7 -> M tile (B = 8*16)
    const int n0    = blockIdx.x << 4;      // hidden-column base
    const int mBase = wave << 4;

    // ---- stage this block's W_hh slice into LDS (16B chunks, 16 per thread)
    for (int idx = threadIdx.x; idx < 64 * (H / 8); idx += 256) {
        const int r = idx >> 6;             // LDS row: gate*16 + col
        const int c = idx & 63;             // 16B chunk within row
        const int gate = r >> 4;
        const int col  = n0 + (r & 15);
        const __bf16* gsrc = Whh + (size_t)(gate * H + col) * H + c * 8;
        __bf16* ldst = smem + r * H + c * 8;
#if ASYNC_LDS
        __builtin_amdgcn_global_load_async_to_lds_b128(
            (g_v4i_ptr)(gsrc), (l_v4i_ptr)(ldst), 0, 0);
#else
        *(v8bf*)ldst = *(const v8bf*)gsrc;
#endif
    }
#if ASYNC_LDS
    asm volatile("s_wait_asynccnt 0" ::: "memory");
#endif
    __syncthreads();

    // ---- prefetch the xg gate rows we will need in the pointwise phase
    {
        const int nPre = n0 + (lane & 15);
        const int mPre = mBase + ((lane & 16) ? 8 : 0);
        const float* gPre = xg + ((size_t)mPre * T + t) * G4H;
        __builtin_prefetch(gPre + 0 * H + nPre, 0, 0);
        __builtin_prefetch(gPre + 1 * H + nPre, 0, 0);
        __builtin_prefetch(gPre + 2 * H + nPre, 0, 0);
        __builtin_prefetch(gPre + 3 * H + nPre, 0, 0);
    }

    // ---- gates GEMM: 4 accumulators (i,f,g,o), B-frags from LDS
    v8f aI = {0.f,0.f,0.f,0.f,0.f,0.f,0.f,0.f};
    v8f aF = aI, aG = aI, aO = aI;
    for (int k = 0; k < H; k += 32) {
        v16bf a  = load_frag_a(h_in, H, mBase, k, lane);
        aI = wmma_bf16(a, load_frag_b_lds(smem,  0, k, lane), aI);
        aF = wmma_bf16(a, load_frag_b_lds(smem, 16, k, lane), aF);
        aG = wmma_bf16(a, load_frag_b_lds(smem, 32, k, lane), aG);
        aO = wmma_bf16(a, load_frag_b_lds(smem, 48, k, lane), aO);
    }

    // ---- LSTM pointwise, entirely in registers (all 4 gates share layout)
    const int n    = n0 + (lane & 15);
    const int mOff = mBase + ((lane & 16) ? 8 : 0);
    #pragma unroll
    for (int i = 0; i < 8; ++i) {
        const int m = mOff + i;
        const float* g = xg + ((size_t)m * T + t) * G4H;
        float iv = sigmoid_(aI[i] + g[0 * H + n]);
        float fv = sigmoid_(aF[i] + g[1 * H + n]);
        float gv = tanh_   (aG[i] + g[2 * H + n]);
        float ov = sigmoid_(aO[i] + g[3 * H + n]);
        const size_t sidx = (size_t)m * H + n;
        float cv = fv * c_state[sidx] + iv * gv;
        c_state[sidx] = cv;
        float hv = ov * tanh_(cv);
        __bf16 hb = (__bf16)hv;
        h_out[sidx] = hb;
        if (h_seq) h_seq[((size_t)m * T + t) * H + n] = hb;
    }
}

// ---------------------------------------------------------------------------
// Classifier: logits = h_last @ Wfc^T + bfc ; log_softmax. Tiny: 1 thread/row.
// ---------------------------------------------------------------------------
__global__ void classifier(const __bf16* __restrict__ h,   // [B][H] bf16
                           const float* __restrict__ Wfc,  // [NC][H] f32
                           const float* __restrict__ bfc,  // [NC]
                           float* __restrict__ out)        // [B][NC]
{
    int b = blockIdx.x * blockDim.x + threadIdx.x;
    if (b >= B) return;
    float logits[NC];
    #pragma unroll
    for (int c = 0; c < NC; ++c) {
        float s = bfc[c];
        const float* w = Wfc + (size_t)c * H;
        const __bf16* hr = h + (size_t)b * H;
        for (int k = 0; k < H; ++k) s += (float)hr[k] * w[k];
        logits[c] = s;
    }
    float mx = logits[0];
    #pragma unroll
    for (int c = 1; c < NC; ++c) mx = fmaxf(mx, logits[c]);
    float sum = 0.f;
    #pragma unroll
    for (int c = 0; c < NC; ++c) sum += __expf(logits[c] - mx);
    float lse = mx + __logf(sum);
    #pragma unroll
    for (int c = 0; c < NC; ++c) out[(size_t)b * NC + c] = logits[c] - lse;
}

// ---------------------------------------------------------------------------
// Host orchestration
// ---------------------------------------------------------------------------
extern "C" void kernel_launch(void* const* d_in, const int* in_sizes, int n_in,
                              void* d_out, int out_size, void* d_ws, size_t ws_size,
                              hipStream_t stream) {
    const float* x    = (const float*)d_in[0];
    const float* Wih0 = (const float*)d_in[1];
    const float* Whh0 = (const float*)d_in[2];
    const float* bih0 = (const float*)d_in[3];
    const float* bhh0 = (const float*)d_in[4];
    const float* Wih1 = (const float*)d_in[5];
    const float* Whh1 = (const float*)d_in[6];
    const float* bih1 = (const float*)d_in[7];
    const float* bhh1 = (const float*)d_in[8];
    const float* Wfc  = (const float*)d_in[9];
    const float* bfc  = (const float*)d_in[10];

    char* wsp = (char*)d_ws;
    auto alloc = [&](size_t bytes) -> char* {
        char* p = wsp;
        wsp += (bytes + 255) & ~(size_t)255;
        return p;
    };

    __bf16* xbf   = (__bf16*)alloc((size_t)B * T * IN * 2);
    __bf16* wih0b = (__bf16*)alloc((size_t)G4H * IN * 2);
    __bf16* whh0b = (__bf16*)alloc((size_t)G4H * H * 2);
    __bf16* wih1b = (__bf16*)alloc((size_t)G4H * H * 2);
    __bf16* whh1b = (__bf16*)alloc((size_t)G4H * H * 2);
    float*  xg    = (float*) alloc((size_t)B * T * G4H * 4);   // reused both layers
    __bf16* hseq  = (__bf16*)alloc((size_t)B * T * H * 2);
    __bf16* hPing = (__bf16*)alloc((size_t)B * H * 2);
    __bf16* hPong = (__bf16*)alloc((size_t)B * H * 2);
    float*  cst   = (float*) alloc((size_t)B * H * 4);

    // --- cast everything the matrix units consume to bf16 (once per call) ---
    auto cast = [&](const float* s, __bf16* d, long n) {
        int grid = (int)((n + 255) / 256);
        if (grid > 4096) grid = 4096;
        cast_f32_bf16<<<grid, 256, 0, stream>>>(s, d, n);
    };
    cast(x,    xbf,   (long)B * T * IN);
    cast(Wih0, wih0b, (long)G4H * IN);
    cast(Whh0, whh0b, (long)G4H * H);
    cast(Wih1, wih1b, (long)G4H * H);
    cast(Whh1, whh1b, (long)G4H * H);

    const size_t stepLds = (size_t)64 * H * sizeof(__bf16);   // 64 KB

    // ---------------- layer 0 ----------------
    // xg = x @ Wih0^T + (b_ih0 + b_hh0)
    gemm_bf16_f32<<<4096, 256, 0, stream>>>(xbf, wih0b, bih0, bhh0, xg,
                                            B * T, G4H, IN);
    init_state<<<(B * H) / 256, 256, 0, stream>>>(hPing, cst, B * H);

    __bf16* hA = hPing;
    __bf16* hB = hPong;
    for (int t = 0; t < T; ++t) {
        lstm_step<<<H / 16, 256, stepLds, stream>>>(xg, whh0b, hA, hB, cst, hseq, t);
        __bf16* tmp = hA; hA = hB; hB = tmp;
    }

    // ---------------- layer 1 ----------------
    gemm_bf16_f32<<<4096, 256, 0, stream>>>(hseq, wih1b, bih1, bhh1, xg,
                                            B * T, G4H, H);
    init_state<<<(B * H) / 256, 256, 0, stream>>>(hA, cst, B * H);

    for (int t = 0; t < T; ++t) {
        lstm_step<<<H / 16, 256, stepLds, stream>>>(xg, whh1b, hA, hB, cst,
                                                    (__bf16*)nullptr, t);
        __bf16* tmp = hA; hA = hB; hB = tmp;
    }

    // ---------------- classifier ----------------
    classifier<<<(B + 63) / 64, 64, 0, stream>>>(hA, Wfc, bfc, (float*)d_out);
}
